// NoisyTopKGating_39745627357691
// MI455X (gfx1250) — compile-verified, matrix operand order checked by reading, not run on hardware
//
#include <hip/hip_runtime.h>
#include <stdint.h>

typedef __attribute__((ext_vector_type(16))) __bf16 v16bf;
typedef __attribute__((ext_vector_type(8)))  float  v8f;
typedef __attribute__((ext_vector_type(4)))  int    v4i;
typedef __attribute__((address_space(1))) v4i g_v4i;   // global int4
typedef __attribute__((address_space(3))) v4i l_v4i;   // LDS int4

#define T_TOK   32768
#define D_DIM   2048
#define E_EXP   64
#define KSTEPS  (D_DIM / 32)    // 64 k-steps of K=32
#define CHUNK   8               // k-steps staged per LDS chunk
#define NCHUNK  (KSTEPS / CHUNK)
#define FRAG_BYTES   1024       // one (fragset, kstep) block: 32 lanes * 32 B
#define CHUNK_BYTES  (8 * CHUNK * FRAG_BYTES)   // 64 KB
#define SMEM_BYTES   (2 * CHUNK_BYTES)          // 128 KB double buffer

#if __has_builtin(__builtin_amdgcn_global_load_async_to_lds_b128) && \
    __has_builtin(__builtin_amdgcn_s_wait_asynccnt)
#define USE_ASYNC 1
#else
#define USE_ASYNC 0
#endif

// Pack two fp32 into one dword of two bf16 (truncation) with a single v_perm_b32.
static __device__ __forceinline__ uint32_t pack2_bf16(float lo, float hi) {
    return __builtin_amdgcn_perm(__float_as_uint(hi), __float_as_uint(lo), 0x07060302u);
}

union AFrag {
    uint32_t u[8];
    v16bf    v;
};

// ---------------------------------------------------------------------------
// Pack kernel: fp32 weights [E=64, D=2048] (gate, noise) -> bf16 B-fragments
// in WMMA lane layout. Fragment f = (fs*KSTEPS + ks), fs = mat*4 + n.
// 32 lanes x 32 bytes each. Total 512 KB in d_ws.
// ---------------------------------------------------------------------------
__global__ __launch_bounds__(256) void pack_weights_kernel(
        const float* __restrict__ w_gate,
        const float* __restrict__ w_noise,
        uint4* __restrict__ outp)
{
    const int id  = blockIdx.x * 256 + threadIdx.x;   // 0 .. 16383
    const int L   = id & 31;
    const int f   = id >> 5;
    const int ks  = f & (KSTEPS - 1);
    const int n   = (f >> 6) & 3;
    const int mat = f >> 8;

    const int e   = n * 16 + (L & 15);
    const int off = (L < 16) ? 0 : 8;
    const int k0  = ks * 32;

    const float* w    = mat ? w_noise : w_gate;
    const float* base = w + (size_t)e * D_DIM + k0 + off;

    uint32_t u[8];
    #pragma unroll
    for (int j = 0; j < 4; ++j)
        u[j] = pack2_bf16(base[2 * j], base[2 * j + 1]);
    #pragma unroll
    for (int j = 0; j < 4; ++j)
        u[4 + j] = pack2_bf16(base[16 + 2 * j], base[16 + 2 * j + 1]);

    outp[(size_t)id * 2 + 0] = make_uint4(u[0], u[1], u[2], u[3]);
    outp[(size_t)id * 2 + 1] = make_uint4(u[4], u[5], u[6], u[7]);
}

// ---------------------------------------------------------------------------
// Cooperative chunk copy: 64 KB of B fragments for k-steps [ks0, ks0+CHUNK)
// into the LDS stage buffer. 128 threads x 32 pieces x 16 B.
// LDS layout: dst[fs*8KB + kk*1KB + lane*32B]  (fs-major, matches reads).
// ---------------------------------------------------------------------------
static __device__ __forceinline__ void copy_chunk(
        const char* __restrict__ gsrc, char* dst, int ks0, int tid)
{
    #pragma unroll
    for (int i = 0; i < 32; ++i) {
        const int c  = tid + 128 * i;          // 16-byte piece index, 0..4095
        const int fs = c >> 9;                 // 512 pieces (8 KB) per fragset
        const int r  = c & 511;
        const char* g = gsrc + (size_t)fs * (KSTEPS * FRAG_BYTES)
                             + (size_t)ks0 * FRAG_BYTES + (size_t)r * 16;
        char* l = dst + (size_t)c * 16;
#if USE_ASYNC
        __builtin_amdgcn_global_load_async_to_lds_b128(
            (g_v4i*)g, (l_v4i*)l, 0, 0);
#else
        *(uint4*)l = *(const uint4*)g;
#endif
    }
}

// ---------------------------------------------------------------------------
// Main fused kernel: 128 threads = 4 waves per block, 128 rows per block.
// B fragments staged in double-buffered LDS via async copy; each wave does
// 2 M-tiles x (gate,noise) x 4 N-tiles = 16 WMMAs per k-step.
// ---------------------------------------------------------------------------
__global__ __launch_bounds__(128) void gating_main_kernel(
        const float* __restrict__ x,
        const float* __restrict__ eps,
        const char*  __restrict__ bw,      // packed weight fragments (512 KB)
        int*   __restrict__ out_idx,       // [T,2]
        float* __restrict__ out_gate)      // [T,2]
{
    extern __shared__ char smem[];         // 128 KB: double-buffered B stage,
                                           // later aliased for logits staging
    const int tid  = threadIdx.x;
    const int wave = tid >> 5;
    const int lane = tid & 31;
    const int l15  = lane & 15;
    const int hi   = lane >> 4;
    const int wg_row0 = blockIdx.x * 128;
    const int row0    = wg_row0 + wave * 32;

    v8f acc[2][2][4];                      // [mtile][matrix][ntile]
    #pragma unroll
    for (int mt = 0; mt < 2; ++mt)
        #pragma unroll
        for (int m = 0; m < 2; ++m)
            #pragma unroll
            for (int n = 0; n < 4; ++n)
                acc[mt][m][n] = (v8f)(0.0f);

    const int off = hi ? 8 : 0;
    const float* xr0 = x + (size_t)(row0 + l15) * D_DIM + off;
    const float* xr1 = xr0 + (size_t)16 * D_DIM;

    // Prefetch chunk 0 into buffer 0.
    copy_chunk(bw, smem, 0, tid);

    for (int ch = 0; ch < NCHUNK; ++ch) {
#if USE_ASYNC
        __builtin_amdgcn_s_wait_asynccnt(0);
#endif
        __syncthreads();                   // chunk ch fully resident
        char* cur = smem + (size_t)(ch & 1) * CHUNK_BYTES;
        if (ch + 1 < NCHUNK)
            copy_chunk(bw, smem + (size_t)((ch + 1) & 1) * CHUNK_BYTES,
                       (ch + 1) * CHUNK, tid);

        const float* x0c = xr0 + (size_t)ch * (CHUNK * 32);
        const float* x1c = xr1 + (size_t)ch * (CHUNK * 32);

        #pragma unroll
        for (int kk = 0; kk < CHUNK; ++kk) {
            // ---- load + convert A fragments (fp32 -> bf16 packed) ---------
            AFrag a0, a1;
            {
                const float4* p0 = (const float4*)(x0c + kk * 32);
                const float4* p1 = (const float4*)(x1c + kk * 32);
                float4 f00 = p0[0], f01 = p0[1], f02 = p0[4], f03 = p0[5];
                float4 f10 = p1[0], f11 = p1[1], f12 = p1[4], f13 = p1[5];
                a0.u[0] = pack2_bf16(f00.x, f00.y); a0.u[1] = pack2_bf16(f00.z, f00.w);
                a0.u[2] = pack2_bf16(f01.x, f01.y); a0.u[3] = pack2_bf16(f01.z, f01.w);
                a0.u[4] = pack2_bf16(f02.x, f02.y); a0.u[5] = pack2_bf16(f02.z, f02.w);
                a0.u[6] = pack2_bf16(f03.x, f03.y); a0.u[7] = pack2_bf16(f03.z, f03.w);
                a1.u[0] = pack2_bf16(f10.x, f10.y); a1.u[1] = pack2_bf16(f10.z, f10.w);
                a1.u[2] = pack2_bf16(f11.x, f11.y); a1.u[3] = pack2_bf16(f11.z, f11.w);
                a1.u[4] = pack2_bf16(f12.x, f12.y); a1.u[5] = pack2_bf16(f12.z, f12.w);
                a1.u[6] = pack2_bf16(f13.x, f13.y); a1.u[7] = pack2_bf16(f13.z, f13.w);
            }

            // ---- 16 WMMAs, B fragments from LDS ---------------------------
            #pragma unroll
            for (int fs = 0; fs < 8; ++fs) {
                const int m = fs >> 2, n = fs & 3;
                v16bf b = *(const v16bf*)(cur + fs * (CHUNK * FRAG_BYTES)
                                              + kk * FRAG_BYTES + lane * 32);
                acc[0][m][n] = __builtin_amdgcn_wmma_f32_16x16x32_bf16(
                    false, a0.v, false, b, (short)0, acc[0][m][n], false, false);
                acc[1][m][n] = __builtin_amdgcn_wmma_f32_16x16x32_bf16(
                    false, a1.v, false, b, (short)0, acc[1][m][n], false, false);
            }
        }
    }

    // ---- epilogue: alias stage LDS as float logits [matrix][row][68] ------
    __syncthreads();                       // all B reads done before aliasing
    float (*lg)[128][68] = (float (*)[128][68])smem;

    #pragma unroll
    for (int mt = 0; mt < 2; ++mt)
        #pragma unroll
        for (int m = 0; m < 2; ++m)
            #pragma unroll
            for (int n = 0; n < 4; ++n)
                #pragma unroll
                for (int i = 0; i < 8; ++i) {
                    const int r = wave * 32 + mt * 16 + hi * 8 + i;
                    lg[m][r][n * 16 + l15] = acc[mt][m][n][i];
                }
    __syncthreads();

    // ---- one thread per token: noisy logits, top-2, softmax ---------------
    const int grow = wg_row0 + tid;
    const float* er = eps + (size_t)grow * E_EXP;

    float best1 = -3.4e38f, best2 = -3.4e38f;
    int   i1 = 0, i2 = 0;
    for (int e = 0; e < E_EXP; ++e) {
        const float c  = lg[0][tid][e];
        const float rn = lg[1][tid][e];
        const float sp = (rn > 20.0f) ? rn : log1pf(__expf(rn));
        const float v  = c + er[e] * sp;
        if (v > best1) { best2 = best1; i2 = i1; best1 = v; i1 = e; }
        else if (v > best2) { best2 = v; i2 = e; }
    }
    const float d  = __expf(best2 - best1);
    const float rs = 1.0f / (1.0f + d);

    out_idx[(size_t)grow * 2 + 0]  = i1;
    out_idx[(size_t)grow * 2 + 1]  = i2;
    out_gate[(size_t)grow * 2 + 0] = rs;
    out_gate[(size_t)grow * 2 + 1] = d * rs;
}

// ---------------------------------------------------------------------------
extern "C" void kernel_launch(void* const* d_in, const int* in_sizes, int n_in,
                              void* d_out, int out_size, void* d_ws, size_t ws_size,
                              hipStream_t stream) {
    const float* x       = (const float*)d_in[0];
    const float* w_gate  = (const float*)d_in[1];
    const float* w_noise = (const float*)d_in[2];
    const float* eps     = (const float*)d_in[3];

    int*   out_idx  = (int*)d_out;                 // first T*2 elements: indices
    float* out_gate = (float*)d_out + T_TOK * 2;   // next  T*2 elements: gates

    // Stage 1: pre-swizzle weights to bf16 WMMA B-fragments (512 KB in d_ws).
    pack_weights_kernel<<<64, 256, 0, stream>>>(w_gate, w_noise, (uint4*)d_ws);

    // Stage 2: fused dual-GEMM + noisy top-2 gating, LDS-staged B.
    gating_main_kernel<<<T_TOK / 128, 128, SMEM_BYTES, stream>>>(
        x, eps, (const char*)d_ws, out_idx, out_gate);
}
